// TransformerWithAttentionLayer_57836029608185
// MI455X (gfx1250) — compile-verified, matrix operand order checked by reading, not run on hardware
//
#include <hip/hip_runtime.h>
#include <hip/hip_bf16.h>

typedef __attribute__((ext_vector_type(16))) _Float16 v16h;
typedef __attribute__((ext_vector_type(8)))  _Float16 v8h;
typedef __attribute__((ext_vector_type(8)))  float    v8f;
typedef _Float16 f16;

#define HW   65536   // 256*256
#define NB   2
#define CIN  64
#define CH   384     // attn 6*dim == ffn hidden*2

// ---------------------------------------------------------------------------
// CDNA5 async global->LDS copy (ASYNCcnt-tracked, ISA 10.x / 15.18.3)
// ---------------------------------------------------------------------------
__device__ __forceinline__ void async_copy_b128(void* lds_dst, const void* gsrc) {
  unsigned lds = (unsigned)(size_t)lds_dst;           // LDS offset taken from the low 32 bits
  unsigned long long ga = (unsigned long long)(size_t)gsrc;
  asm volatile("global_load_async_to_lds_b128 %0, %1, off"
               :: "v"(lds), "v"(ga) : "memory");
}
__device__ __forceinline__ void async_wait0() {
  asm volatile("s_wait_asynccnt 0x0" ::: "memory");
}

// ---------------------------------------------------------------------------
// WMMA helpers (CDNA5 16x16x32 f16 -> f32 accumulate)
// ---------------------------------------------------------------------------
__device__ __forceinline__ v8f wmma_f16(v16h a, v16h b, v8f c) {
  return __builtin_amdgcn_wmma_f32_16x16x32_f16(false, a, false, b, (short)0, c,
                                                false, false);
}

// 16-bit A/B fragment layout (ISA 7.12.2): lane<16 holds K {0..7,16..23},
// lane>=16 holds K {8..15,24..31}; row (M or N) = lane%16.
__device__ __forceinline__ v16h frag_lds(const f16* base, int row, int stride,
                                         int kbase, int khalf) {
  const f16* p = base + row * stride + kbase + khalf * 8;
  v16h f;
#pragma unroll
  for (int e = 0; e < 8; ++e) f[e] = p[e];
#pragma unroll
  for (int e = 0; e < 8; ++e) f[8 + e] = p[16 + e];
  return f;
}

__device__ __forceinline__ v16h frag_gw(const float* w, int row, int stride,
                                        int kbase, int khalf) {
  const float* p = w + (size_t)row * stride + kbase + khalf * 8;
  v16h f;
#pragma unroll
  for (int e = 0; e < 8; ++e) f[e] = (f16)p[e];
#pragma unroll
  for (int e = 0; e < 8; ++e) f[8 + e] = (f16)p[16 + e];
  return f;
}

// ---------------------------------------------------------------------------
// K0: spatial kernels g[c][8][8] = irfft2(fft_params[c])  (real spectrum)
// ---------------------------------------------------------------------------
__global__ void k_spectral(const float* __restrict__ fp, float* __restrict__ g) {
  int c = blockIdx.x, p = threadIdx.x;
  int m = p >> 3, n = p & 7;
  float acc = 0.f;
  for (int ky = 0; ky < 8; ++ky)
    for (int kx = 0; kx < 5; ++kx) {
      float wkx = (kx == 0 || kx == 4) ? 1.f : 2.f;  // hermitian fold weight
      float ang = 0.78539816339f * (float)(ky * m + kx * n);  // 2*pi/8
      acc += wkx * fp[c * 40 + ky * 5 + kx] * cosf(ang);
    }
  g[c * 64 + p] = acc * (1.f / 64.f);
}

// ---------------------------------------------------------------------------
// K1: fused segment-LayerNorm + 1x1 conv (64 -> 384) via WMMA.
// One block = 128 consecutive pixels (2 LN segments per channel).
// x tile staged into LDS with async global->LDS copies.
// ---------------------------------------------------------------------------
__global__ __launch_bounds__(256)
void k_ln_pin(const float* __restrict__ xin, const float* __restrict__ nw,
              const float* __restrict__ nbias, const float* __restrict__ pw,
              f16* __restrict__ hout) {
  __shared__ float XT[64 * 128];  // 32 KB, [channel][pixel]
  __shared__ f16 A[128 * 64];     // 16 KB, [pixel][channel] f16
  int t = threadIdx.x;
  int b = blockIdx.x >> 9;        // 512 blocks per batch image
  int hwb = (blockIdx.x & 511) * 128;

  // pull the 96KB weight matrix through L2/L1 while staging runs
  if (t < 96) __builtin_prefetch(pw + t * 256, 0, 3);

  // async-stage the 64ch x 128px f32 tile (2048 x b128)
#pragma unroll
  for (int i = 0; i < 8; ++i) {
    int j = t + i * 256;          // b128 chunk id
    int c = j >> 5, ck = j & 31;  // 32 chunks of 4 floats per channel row
    async_copy_b128(&XT[c * 128 + ck * 4],
                    xin + ((size_t)b * CIN + c) * HW + hwb + ck * 4);
  }
  async_wait0();
  __syncthreads();

  if (t < 128) {                  // one thread per (channel, segment)
    int c = t >> 1, sh = t & 1;
    const float* px = &XT[c * 128 + sh * 64];
    float s = 0.f, ss = 0.f;
#pragma unroll 8
    for (int i = 0; i < 64; ++i) { float v = px[i]; s += v; ss += v * v; }
    float mu = s * (1.f / 64.f);
    float inv = rsqrtf(ss * (1.f / 64.f) - mu * mu + 1e-5f);
#pragma unroll 8
    for (int i = 0; i < 64; ++i) {
      float v = (px[i] - mu) * inv * nw[i] + nbias[i];
      A[(sh * 64 + i) * 64 + c] = (f16)v;
    }
  }
  __syncthreads();

  int wave = t >> 5, lane = t & 31;
  int lrow = lane & 15, kh = lane >> 4;
  int ocb = wave * 48;            // each wave: 48 out-channels x 128 pixels
  v16h bf[3][2];
#pragma unroll
  for (int nt = 0; nt < 3; ++nt)
#pragma unroll
    for (int ks = 0; ks < 2; ++ks)
      bf[nt][ks] = frag_gw(pw, ocb + nt * 16 + lrow, 64, ks * 32, kh);

  for (int mt = 0; mt < 8; ++mt) {
    v16h a0 = frag_lds(A, mt * 16 + lrow, 64, 0, kh);
    v16h a1 = frag_lds(A, mt * 16 + lrow, 64, 32, kh);
#pragma unroll
    for (int nt = 0; nt < 3; ++nt) {
      v8f acc = {};
      acc = wmma_f16(a0, bf[nt][0], acc);
      acc = wmma_f16(a1, bf[nt][1], acc);
      // D layout: lane holds fixed N=oc, 8 consecutive pixels -> b128 store
      int oc = ocb + nt * 16 + lrow;
      int m0 = mt * 16 + kh * 8;
      f16* po = hout + ((size_t)b * CH + oc) * HW + hwb + m0;
      v8h v;
#pragma unroll
      for (int r = 0; r < 8; ++r) v[r] = (f16)acc[r];
      *(v8h*)po = v;
    }
  }
}

// ---------------------------------------------------------------------------
// K2: attn tail: dwconv3x3 -> q(*)k 8x8 circular conv -> *v -> pout GEMM
// (K=128 -> 64) + residual.  One block = one 64-pixel segment, all channels.
// Circular conv is register-tiled: thread = (channel, output row); q/k rows
// come in as single ds_load_b128's, the 8x8 tap block is all-register FMA
// with compile-time rotation indices.
// ---------------------------------------------------------------------------
__global__ __launch_bounds__(256)
void k_attn_tail(const f16* __restrict__ h, const float* __restrict__ dww,
                 const float* __restrict__ pw, const float* __restrict__ xin,
                 float* __restrict__ y1) {
  __shared__ f16 QKV[CH * 64];      // 48 KB, [channel][pixel]
  int t = threadIdx.x;
  int b = blockIdx.x >> 10;
  int hwb = (blockIdx.x & 1023) * 64;

  if (t < 32) __builtin_prefetch(pw + t * 1024, 0, 3);

  // phase 1: depthwise 3x3 (zero pad), input f16 L2-resident
  for (int it = 0; it < 96; ++it) {
    int j = t + it * 256;           // 384ch * 64px
    int c = j >> 6, p = j & 63;
    int hw = hwb + p;
    int hr = hw >> 8, wc = hw & 255;
    const f16* hc = h + ((size_t)b * CH + c) * HW;
    const float* wk = dww + c * 9;
    float acc = 0.f;
#pragma unroll
    for (int dy = 0; dy < 3; ++dy) {
      int r = hr + dy - 1;
      if (r < 0 || r >= 256) continue;
#pragma unroll
      for (int dx = 0; dx < 3; ++dx) {
        int cc = wc + dx - 1;
        if (cc < 0 || cc >= 256) continue;
        acc += (float)hc[r * 256 + cc] * wk[dy * 3 + dx];
      }
    }
    QKV[c * 64 + p] = (f16)acc;
  }
  __syncthreads();

  // phase 2: per-channel 8x8 circular conv, register-tiled (8 outputs/thread)
  float u[4][8];
  for (int it = 0; it < 4; ++it) {
    int j = t + it * 256;           // 128ch * 8 output rows
    int c = j >> 3, pm = j & 7;
    const f16* q = &QKV[c * 64];
    const f16* kk = &QKV[(c + 128) * 64];
    float acc[8];
#pragma unroll
    for (int pn = 0; pn < 8; ++pn) acc[pn] = 0.f;
#pragma unroll
    for (int tm = 0; tm < 8; ++tm) {
      int rm = ((pm - tm) & 7) << 3;
      float qr[8], kr[8];
#pragma unroll
      for (int e = 0; e < 8; ++e) qr[e] = (float)q[tm * 8 + e];
#pragma unroll
      for (int e = 0; e < 8; ++e) kr[e] = (float)kk[rm + e];
#pragma unroll
      for (int tn = 0; tn < 8; ++tn)
#pragma unroll
        for (int pn = 0; pn < 8; ++pn)
          acc[pn] += qr[tn] * kr[(pn - tn) & 7];   // compile-time index
    }
    const f16* v = &QKV[(c + 256) * 64 + pm * 8];
#pragma unroll
    for (int pn = 0; pn < 8; ++pn) u[it][pn] = (float)v[pn] * acc[pn];
  }
  __syncthreads();

  // phase 3: repack u -> A [pixel][128ch] (reuse LDS)
  f16* A = QKV;
  for (int it = 0; it < 4; ++it) {
    int j = t + it * 256;
    int c = j >> 3, pm = j & 7;
#pragma unroll
    for (int pn = 0; pn < 8; ++pn) A[(pm * 8 + pn) * 128 + c] = (f16)u[it][pn];
  }
  __syncthreads();

  // phase 4: pout GEMM M=64 N=64 K=128 (+ residual x)
  int wave = t >> 5, lane = t & 31;
  int lrow = lane & 15, kh = lane >> 4;
  for (int pr = 0; pr < 2; ++pr) {
    int idx = wave * 2 + pr;
    int mt = idx >> 2, nt = idx & 3;
    v8f acc = {};
#pragma unroll
    for (int ks = 0; ks < 4; ++ks) {
      v16h a = frag_lds(A, mt * 16 + lrow, 128, ks * 32, kh);
      v16h bb = frag_gw(pw, nt * 16 + lrow, 128, ks * 32, kh);
      acc = wmma_f16(a, bb, acc);
    }
    int oc = nt * 16 + lrow;
    int m0 = mt * 16 + kh * 8;
    const float* px = xin + ((size_t)b * CIN + oc) * HW + hwb + m0;
    float* po = y1 + ((size_t)b * CIN + oc) * HW + hwb + m0;
#pragma unroll
    for (int r = 0; r < 8; ++r) po[r] = px[r] + acc[r];
  }
}

// ---------------------------------------------------------------------------
// K3b: in-place per-segment circular conv with per-channel kernel g
// (the DFFN spectral multiply).  Block = (segment, 128-channel group).
// Tile staged with async global->LDS; conv register-tiled as in K2.
// ---------------------------------------------------------------------------
__global__ __launch_bounds__(256)
void k_patchconv(f16* __restrict__ h2, const float* __restrict__ g) {
  __shared__ f16 S[128 * 64];       // 16 KB
  int t = threadIdx.x;
  int seg = blockIdx.x / 3, cg = blockIdx.x % 3;
  int b = seg >> 10;
  int hwb = (seg & 1023) * 64;
  int cb = cg * 128;

  // async-stage 128ch x 64px f16 (1024 x b128)
#pragma unroll
  for (int i = 0; i < 4; ++i) {
    int j8 = (t + i * 256) * 8;     // element index, 8 f16 per transfer
    int c = j8 >> 6, p = j8 & 63;
    async_copy_b128(&S[j8], &h2[((size_t)b * CH + cb + c) * HW + hwb + p]);
  }
  async_wait0();
  __syncthreads();

  for (int it = 0; it < 4; ++it) {
    int j = t + it * 256;           // 128ch * 8 output rows
    int c = j >> 3, pm = j & 7;
    const f16* sc = &S[c * 64];
    const float* gc = g + (cb + c) * 64;
    float acc[8];
#pragma unroll
    for (int pn = 0; pn < 8; ++pn) acc[pn] = 0.f;
#pragma unroll
    for (int tm = 0; tm < 8; ++tm) {
      int rm = ((pm - tm) & 7) << 3;
      float sr[8], gr[8];
#pragma unroll
      for (int e = 0; e < 8; ++e) sr[e] = (float)sc[tm * 8 + e];
#pragma unroll
      for (int e = 0; e < 8; ++e) gr[e] = gc[rm + e];
#pragma unroll
      for (int tn = 0; tn < 8; ++tn)
#pragma unroll
        for (int pn = 0; pn < 8; ++pn)
          acc[pn] += sr[tn] * gr[(pn - tn) & 7];
    }
    f16* po = &h2[((size_t)b * CH + cb + c) * HW + hwb + pm * 8];
#pragma unroll
    for (int pn = 0; pn < 8; ++pn) po[pn] = (f16)acc[pn];
  }
}

// ---------------------------------------------------------------------------
// K4: ffn tail: dwconv3x3 -> gelu(x1)*x2 -> pout GEMM (K=192 -> 64) + residual
// ---------------------------------------------------------------------------
__global__ __launch_bounds__(256)
void k_ffn_tail(const f16* __restrict__ p2, const float* __restrict__ dww,
                const float* __restrict__ pw, const float* __restrict__ y1,
                float* __restrict__ out) {
  __shared__ f16 DW[CH * 64];       // 48 KB
  int t = threadIdx.x;
  int b = blockIdx.x >> 10;
  int hwb = (blockIdx.x & 1023) * 64;

  if (t < 48) __builtin_prefetch(pw + t * 1024, 0, 3);

  for (int it = 0; it < 96; ++it) { // dwconv3x3 on 384 channels
    int j = t + it * 256;
    int c = j >> 6, p = j & 63;
    int hw = hwb + p;
    int hr = hw >> 8, wc = hw & 255;
    const f16* hc = p2 + ((size_t)b * CH + c) * HW;
    const float* wk = dww + c * 9;
    float acc = 0.f;
#pragma unroll
    for (int dy = 0; dy < 3; ++dy) {
      int r = hr + dy - 1;
      if (r < 0 || r >= 256) continue;
#pragma unroll
      for (int dx = 0; dx < 3; ++dx) {
        int cc = wc + dx - 1;
        if (cc < 0 || cc >= 256) continue;
        acc += (float)hc[r * 256 + cc] * wk[dy * 3 + dx];
      }
    }
    DW[c * 64 + p] = (f16)acc;
  }
  __syncthreads();

  float u[48];                      // gelu(x1) * x2, 192 channels
  for (int it = 0; it < 48; ++it) {
    int j = t + it * 256;
    int c = j / 64, p = j & 63;
    float x1 = (float)DW[c * 64 + p];
    float x2 = (float)DW[(c + 192) * 64 + p];
    float ge = 0.5f * x1 * (1.f + erff(x1 * 0.70710678118f));  // exact gelu
    u[it] = ge * x2;
  }
  __syncthreads();

  f16* A = DW;                      // repack -> [pixel][192ch]
  for (int it = 0; it < 48; ++it) {
    int j = t + it * 256;
    int c = j / 64, p = j & 63;
    A[p * 192 + c] = (f16)u[it];
  }
  __syncthreads();

  int wave = t >> 5, lane = t & 31;
  int lrow = lane & 15, kh = lane >> 4;
  for (int pr = 0; pr < 2; ++pr) {
    int idx = wave * 2 + pr;
    int mt = idx >> 2, nt = idx & 3;
    v8f acc = {};
#pragma unroll
    for (int ks = 0; ks < 6; ++ks) {  // K = 192
      v16h a = frag_lds(A, mt * 16 + lrow, 192, ks * 32, kh);
      v16h bb = frag_gw(pw, nt * 16 + lrow, 192, ks * 32, kh);
      acc = wmma_f16(a, bb, acc);
    }
    int oc = nt * 16 + lrow;
    int m0 = mt * 16 + kh * 8;
    const float* pr1 = y1 + ((size_t)b * CIN + oc) * HW + hwb + m0;
    float* po = out + ((size_t)b * CIN + oc) * HW + hwb + m0;
#pragma unroll
    for (int r = 0; r < 8; ++r) po[r] = pr1[r] + acc[r];
  }
}

// ---------------------------------------------------------------------------
extern "C" void kernel_launch(void* const* d_in, const int* in_sizes, int n_in,
                              void* d_out, int out_size, void* d_ws,
                              size_t ws_size, hipStream_t stream) {
  const float* x     = (const float*)d_in[0];
  const float* n1w   = (const float*)d_in[1];
  const float* n1b   = (const float*)d_in[2];
  const float* apin  = (const float*)d_in[3];
  const float* adw   = (const float*)d_in[4];
  const float* apout = (const float*)d_in[5];
  const float* n2w   = (const float*)d_in[6];
  const float* n2b   = (const float*)d_in[7];
  const float* fpin  = (const float*)d_in[8];
  const float* fdw   = (const float*)d_in[9];
  const float* ffp   = (const float*)d_in[10];
  const float* fpout = (const float*)d_in[11];
  float* out = (float*)d_out;

  // workspace: h (f16 NB*CH*HW = 100.7 MB, L2-resident) | y1 (f32 33.5 MB) | g
  char* ws = (char*)d_ws;
  f16*   h  = (f16*)ws;
  float* y1 = (float*)(ws + (size_t)NB * CH * HW * sizeof(f16));
  float* g  = (float*)(ws + (size_t)NB * CH * HW * sizeof(f16)
                          + (size_t)NB * CIN * HW * sizeof(float));

  k_spectral <<<CH,   64,  0, stream>>>(ffp, g);
  k_ln_pin   <<<1024, 256, 0, stream>>>(x, n1w, n1b, apin, h);      // LN1+pin
  k_attn_tail<<<2048, 256, 0, stream>>>(h, adw, apout, x, y1);      // FSAS rest
  k_ln_pin   <<<1024, 256, 0, stream>>>(y1, n2w, n2b, fpin, h);     // LN2+pin
  k_patchconv<<<6144, 256, 0, stream>>>(h, g);                      // spectral
  k_ffn_tail <<<2048, 256, 0, stream>>>(h, fdw, fpout, y1, out);    // DFFN rest
}
